// GraphAutoencoder_77129022701569
// MI455X (gfx1250) — matrix operand order repaired and checked by reading, not compiled, not run on hardware
//
#include <hip/hip_runtime.h>
#include <hip/hip_bf16.h>

typedef __attribute__((ext_vector_type(16))) __bf16 bf16x16;
typedef __attribute__((ext_vector_type(8)))  __bf16 bf16x8;
typedef __attribute__((ext_vector_type(8)))  float  f32x8;

#define N_NODES 16384
#define IN_SZ   4096
#define EMB     500
#define EMB_P   512   // padded embedding (K/N dim for WMMA)

// Guaranteed single-instruction f32 atomic add (no CAS loop).
__device__ __forceinline__ void atomic_add_f32(float* p, float v) {
  asm volatile("global_atomic_add_f32 %0, %1, off" : : "v"(p), "v"(v) : "memory");
}

// ---------------- zero fill (aggr + deg, contiguous in ws) ----------------
__global__ void gae_zero_kernel(float4* __restrict__ p, int n4) {
  int i = blockIdx.x * blockDim.x + threadIdx.x;
  if (i < n4) p[i] = make_float4(0.f, 0.f, 0.f, 0.f);
}

// ---------------- degree: deg[row[e]] += 1 ----------------
__global__ void gae_degree_kernel(const int* __restrict__ row, float* __restrict__ deg, int E) {
  int e = blockIdx.x * blockDim.x + threadIdx.x;
  if (e < E) atomic_add_f32(&deg[row[e]], 1.0f);
}

// ---------------- dinv = deg>0 ? rsqrt(deg) : 0 (in place) ----------------
__global__ void gae_dinv_kernel(float* __restrict__ deg) {
  int i = blockIdx.x * blockDim.x + threadIdx.x;
  float d = deg[i];
  deg[i] = (d > 0.f) ? rsqrtf(d) : 0.f;
}

// ---------------- edge scatter: one wave32 per edge ----------------
__global__ __launch_bounds__(256)
void gae_scatter_kernel(const int* __restrict__ ei, const float* __restrict__ h,
                        const float* __restrict__ dinv, const float* __restrict__ cw,
                        float* __restrict__ aggr, int E) {
  int gw   = (blockIdx.x * blockDim.x + threadIdx.x) >> 5;   // global wave id == edge id
  int lane = threadIdx.x & 31;
  if (gw >= E) return;
  int r = ei[gw];
  int c = ei[E + gw];
  float norm = dinv[r] * dinv[c] * cw[0];
  const float* hr = h    + (size_t)r * EMB_P;
  float*       ac = aggr + (size_t)c * EMB_P;
#pragma unroll
  for (int i = 0; i < EMB_P / 32; ++i) {
    int col = lane + i * 32;
    atomic_add_f32(&ac[col], norm * hr[col]);   // aggr is L2-resident (32 MB)
  }
}

// ---------------- shared WMMA compute: 32x32 sub-tile per wave ----------------
__device__ __forceinline__ void wmma_tile(const __bf16 (&As)[128][32], const __bf16 (&Bs)[64][32],
                                          f32x8 (&acc)[2][2], int wm, int wn, int hlf, int lr) {
  bf16x16 afrag[2], bfrag[2];
#pragma unroll
  for (int i = 0; i < 2; ++i) {          // A: lane-half picks K {0-7,16-23} vs {8-15,24-31}
    int row = wm * 32 + i * 16 + lr;
    bf16x8 lo = *reinterpret_cast<const bf16x8*>(&As[row][hlf * 8]);
    bf16x8 hi = *reinterpret_cast<const bf16x8*>(&As[row][16 + hlf * 8]);
    afrag[i] = __builtin_shufflevector(lo, hi, 0,1,2,3,4,5,6,7,8,9,10,11,12,13,14,15);
  }
#pragma unroll
  for (int j = 0; j < 2; ++j) {          // B: lane-half picks K {0-7,16-23} vs {8-15,24-31}
    int col = wn * 32 + j * 16 + lr;
    bf16x8 lo = *reinterpret_cast<const bf16x8*>(&Bs[col][hlf * 8]);
    bf16x8 hi = *reinterpret_cast<const bf16x8*>(&Bs[col][16 + hlf * 8]);
    bfrag[j] = __builtin_shufflevector(lo, hi, 0,1,2,3,4,5,6,7,8,9,10,11,12,13,14,15);
  }
#pragma unroll
  for (int i = 0; i < 2; ++i)
#pragma unroll
    for (int j = 0; j < 2; ++j)
      acc[i][j] = __builtin_amdgcn_wmma_f32_16x16x32_bf16(
          false, afrag[i], false, bfrag[j], (short)0, acc[i][j], false, false);
}

__device__ __forceinline__ void cvt_store8(__bf16* dst, float4 v0, float4 v1) {
  dst[0] = (__bf16)v0.x; dst[1] = (__bf16)v0.y; dst[2] = (__bf16)v0.z; dst[3] = (__bf16)v0.w;
  dst[4] = (__bf16)v1.x; dst[5] = (__bf16)v1.y; dst[6] = (__bf16)v1.z; dst[7] = (__bf16)v1.w;
}

// ---------------- encode GEMM: h = sigmoid(X @ Wenc^T + benc) ----------------
// block = 256 thr (8 waves); tile = 128(M) x 64(N); K step 32; bf16 WMMA, f32 acc.
// Software-pipelined: global loads of tile t+1 issue before WMMA of tile t;
// double-buffered LDS, one barrier per K-step.
__global__ __launch_bounds__(256)
void gae_encode_kernel(const float* __restrict__ x, const float* __restrict__ wenc,
                       const float* __restrict__ benc, float* __restrict__ h) {
  __shared__ __align__(16) __bf16 As[2][128][32];   // A tile, row-major
  __shared__ __align__(16) __bf16 Bs[2][64][32];    // B tile, stored [n][k]

  const int mbase = blockIdx.y * 128;
  const int nbase = blockIdx.x * 64;
  const int t    = threadIdx.x;
  const int wave = t >> 5, lane = t & 31;
  const int wm = wave & 3, wn = wave >> 2;          // 4x2 wave grid of 32x32 sub-tiles
  const int hlf = lane >> 4, lr = lane & 15;

  f32x8 acc[2][2] = {};

  const int arow = t >> 1, ahalf = t & 1;           // A staging: 16 floats per thread
  const int bn = t >> 2,  bkk = (t & 3) * 8;        // B staging: 8 floats per thread
  const int gbn = nbase + bn;

  float4 ra[4], rb[2];

  auto load_tile = [&](int k0) {
    const float4* sa = reinterpret_cast<const float4*>(
        x + (size_t)(mbase + arow) * IN_SZ + k0 + ahalf * 16);
#pragma unroll
    for (int q = 0; q < 4; ++q) ra[q] = sa[q];
    if (k0 + 32 < IN_SZ) __builtin_prefetch(sa + 8, 0, 1);   // next A tile
    if (gbn < EMB) {
      const float4* sb = reinterpret_cast<const float4*>(
          wenc + (size_t)gbn * IN_SZ + k0 + bkk);
      rb[0] = sb[0]; rb[1] = sb[1];
    } else {
      rb[0] = rb[1] = make_float4(0.f, 0.f, 0.f, 0.f);
    }
  };
  auto store_tile = [&](int p) {
    __bf16* da = &As[p][arow][ahalf * 16];
    cvt_store8(da,     ra[0], ra[1]);
    cvt_store8(da + 8, ra[2], ra[3]);
    cvt_store8(&Bs[p][bn][bkk], rb[0], rb[1]);
  };

  load_tile(0);
  store_tile(0);
  __syncthreads();
  int p = 0;
  for (int k0 = 32; k0 < IN_SZ; k0 += 32) {
    load_tile(k0);                       // global loads in flight...
    wmma_tile(As[p], Bs[p], acc, wm, wn, hlf, lr);   // ...while WMMA runs
    store_tile(p ^ 1);
    __syncthreads();
    p ^= 1;
  }
  wmma_tile(As[p], Bs[p], acc, wm, wn, hlf, lr);

  // epilogue: bias + sigmoid
#pragma unroll
  for (int j = 0; j < 2; ++j) {
    int gcol = nbase + wn * 32 + j * 16 + lr;
    float bias = (gcol < EMB) ? benc[gcol] : 0.f;
#pragma unroll
    for (int i = 0; i < 2; ++i) {
      int rbase = mbase + wm * 32 + i * 16 + hlf * 8;
#pragma unroll
      for (int e = 0; e < 8; ++e) {
        float v = acc[i][j][e] + bias;
        h[(size_t)(rbase + e) * EMB_P + gcol] = 1.f / (1.f + __expf(-v));
      }
    }
  }
}

// ---------------- decode GEMM: out = (aggr + h) @ Wdec^T + bdec ----------------
__global__ __launch_bounds__(256)
void gae_decode_kernel(const float* __restrict__ aggr, const float* __restrict__ h,
                       const float* __restrict__ wdec, const float* __restrict__ bdec,
                       float* __restrict__ out) {
  __shared__ __align__(16) __bf16 As[2][128][32];
  __shared__ __align__(16) __bf16 Bs[2][64][32];

  const int mbase = blockIdx.y * 128;
  const int nbase = blockIdx.x * 64;
  const int t    = threadIdx.x;
  const int wave = t >> 5, lane = t & 31;
  const int wm = wave & 3, wn = wave >> 2;
  const int hlf = lane >> 4, lr = lane & 15;

  f32x8 acc[2][2] = {};

  const int arow = t >> 1, ahalf = t & 1;
  const int bn = t >> 2,  bkk = (t & 3) * 8;
  const int gbn = nbase + bn;            // output column (0..4095)

  float4 ra[4], rh[4];
  float  rbv[8];

  auto load_tile = [&](int k0) {
    size_t base = (size_t)(mbase + arow) * EMB_P + k0 + ahalf * 16;
    const float4* sa = reinterpret_cast<const float4*>(aggr + base);
    const float4* sh = reinterpret_cast<const float4*>(h + base);
#pragma unroll
    for (int q = 0; q < 4; ++q) { ra[q] = sa[q]; rh[q] = sh[q]; }
    const float* sb = wdec + (size_t)gbn * EMB;
#pragma unroll
    for (int q = 0; q < 8; ++q) {
      int gk = k0 + bkk + q;
      rbv[q] = (gk < EMB) ? sb[gk] : 0.f;
    }
  };
  auto store_tile = [&](int p) {
    __bf16* da = &As[p][arow][ahalf * 16];
#pragma unroll
    for (int q = 0; q < 4; ++q) {
      da[4*q+0] = (__bf16)(ra[q].x + rh[q].x);
      da[4*q+1] = (__bf16)(ra[q].y + rh[q].y);
      da[4*q+2] = (__bf16)(ra[q].z + rh[q].z);
      da[4*q+3] = (__bf16)(ra[q].w + rh[q].w);
    }
    __bf16* db = &Bs[p][bn][bkk];
#pragma unroll
    for (int q = 0; q < 8; ++q) db[q] = (__bf16)rbv[q];
  };

  load_tile(0);
  store_tile(0);
  __syncthreads();
  int p = 0;
  for (int k0 = 32; k0 < EMB_P; k0 += 32) {
    load_tile(k0);
    wmma_tile(As[p], Bs[p], acc, wm, wn, hlf, lr);
    store_tile(p ^ 1);
    __syncthreads();
    p ^= 1;
  }
  wmma_tile(As[p], Bs[p], acc, wm, wn, hlf, lr);

  // epilogue: bias, write fp32 output
#pragma unroll
  for (int j = 0; j < 2; ++j) {
    int gcol = nbase + wn * 32 + j * 16 + lr;
    float bias = bdec[gcol];
#pragma unroll
    for (int i = 0; i < 2; ++i) {
      int rbase = mbase + wm * 32 + i * 16 + hlf * 8;
#pragma unroll
      for (int e = 0; e < 8; ++e) {
        out[(size_t)(rbase + e) * IN_SZ + gcol] = acc[i][j][e] + bias;
      }
    }
  }
}

extern "C" void kernel_launch(void* const* d_in, const int* in_sizes, int n_in,
                              void* d_out, int out_size, void* d_ws, size_t ws_size,
                              hipStream_t stream) {
  const float* x    = (const float*)d_in[0];
  const int*   ei   = (const int*)d_in[1];
  const float* wenc = (const float*)d_in[2];
  const float* benc = (const float*)d_in[3];
  const float* wdec = (const float*)d_in[4];
  const float* bdec = (const float*)d_in[5];
  const float* cw   = (const float*)d_in[6];
  float* out = (float*)d_out;
  const int E = in_sizes[1] / 2;

  // workspace layout: h [32MB] | aggr [32MB] | deg [64KB]
  char* ws = (char*)d_ws;
  float* h    = (float*)(ws);
  float* aggr = (float*)(ws + (size_t)N_NODES * EMB_P * 4);
  float* deg  = (float*)(ws + (size_t)N_NODES * EMB_P * 8);

  // zero aggr + deg (contiguous) every call — graph replays don't re-poison
  int zero4 = (N_NODES * EMB_P + N_NODES) / 4;
  gae_zero_kernel<<<(zero4 + 255) / 256, 256, 0, stream>>>((float4*)aggr, zero4);

  // encode: h = sigmoid(x @ wenc^T + benc)
  dim3 egrid(EMB_P / 64, N_NODES / 128);
  gae_encode_kernel<<<egrid, 256, 0, stream>>>(x, wenc, benc, h);

  // graph conv normalization
  gae_degree_kernel<<<(E + 255) / 256, 256, 0, stream>>>(ei, deg, E);
  gae_dinv_kernel<<<N_NODES / 256, 256, 0, stream>>>(deg);

  // message scatter: aggr[col] += norm * w * h[row]
  gae_scatter_kernel<<<(E * 32) / 256, 256, 0, stream>>>(ei, h, deg, cw, aggr, E);

  // decode: out = (aggr + h) @ wdec^T + bdec
  dim3 dgrid(IN_SZ / 64, N_NODES / 128);
  gae_decode_kernel<<<dgrid, 256, 0, stream>>>(aggr, h, wdec, bdec, out);
}
